// PatchDecoder_74088185856599
// MI455X (gfx1250) — matrix operand order repaired and checked by reading, not compiled, not run on hardware
//
#include <hip/hip_runtime.h>
#include <hip/hip_bf16.h>
#include <math.h>

// ---------------------------------------------------------------------------
// Problem constants (from reference)
// ---------------------------------------------------------------------------
#define B_      32
#define K_      16
#define P_      1024
#define DSLOT_  256
#define DIN_    256
#define H_      1024
#define M_      768
#define TOPK_   4

// WMMA tiling: 16x16x32 bf16 -> f32
#define NT1 64          // H/16
#define NT2 64          // H/16
#define NT3 49          // ceil(769/16) -> 784 padded cols
#define KC1 8           // DIN/32
#define KC2 32          // H/32
#define KC3 32          // H/32

// Fused-kernel tile: 64 rows = 16 positions x TOPK, 512 threads = 16 wave32.
#define ROWS_   64
#define NPOS_   16
#define THREADS_ 512
#define NWAVES_ 16

// LDS layout (bytes). Buffers aliased by lifetime:
//   bufA: xs (phase0-1, 64x264 bf16 = 33792) OR h2 (phase2-3, 64x1040 bf16)
//   bufB: h1 (phase1-2, 64x1040 bf16)        OR outs bf16 (phase3-4, 64x784)
#define XS_LD    264
#define H_LD     1040
#define OUT_LD   784
#define BUFA_BYTES 133120   // 64*1040*2
#define BUFB_BYTES 133120   // 64*1040*2 >= 64*784*2
// extras: rowSlot(256) + alphas(256) + wgt(256)

typedef __bf16 bf16_t;
typedef __bf16 v16bf __attribute__((ext_vector_type(16)));
typedef __bf16 v8bf  __attribute__((ext_vector_type(8)));
typedef __bf16 v2bf  __attribute__((ext_vector_type(2)));
typedef float  v8f   __attribute__((ext_vector_type(8)));

#define WMMA_BF16(a, b, c) \
    __builtin_amdgcn_wmma_f32_16x16x32_bf16(false, (a), false, (b), (short)0, (c), false, false)

// ---------------------------------------------------------------------------
// Weight packing: row-major fp32 W[Krows x N] -> bf16 WMMA B-fragments.
// Fragment (kc,nt): 32 lanes x 16 bf16; lane holds column nt*16+(lane&15),
// element e is K-row kc*32 + 16*(e>>3) + 8*(lane>>4) + (e&7)  (CDNA5 16-bit B layout).
// One contiguous 32B load per lane at runtime.
// ---------------------------------------------------------------------------
__global__ void pack_weights_kernel(const float* __restrict__ W,
                                    bf16_t* __restrict__ Wp,
                                    int Krows, int N, int NT, int total) {
    int idx = blockIdx.x * blockDim.x + threadIdx.x;
    if (idx >= total) return;
    int kc   = idx / (NT * 512);
    int rem  = idx - kc * (NT * 512);
    int nt   = rem >> 9;
    int q    = rem & 511;
    int lane = q >> 4;
    int e    = q & 15;
    int krow = kc * 32 + ((e >> 3) << 4) + ((lane >> 4) << 3) + (e & 7);
    int col  = nt * 16 + (lane & 15);
    float v = (krow < Krows && col < N) ? W[krow * N + col] : 0.0f;
    Wp[idx] = (bf16_t)v;
}

// ---------------------------------------------------------------------------
// slot_proj[b*K+k][d] = sum_s slots[b,k,s] * Wt[s,d] + bt[d]   (tiny, scalar)
// ---------------------------------------------------------------------------
__global__ void slot_proj_kernel(const float* __restrict__ slots,
                                 const float* __restrict__ Wt,
                                 const float* __restrict__ bt,
                                 float* __restrict__ sp) {
    int row = blockIdx.x;        // 0..B*K-1
    int col = threadIdx.x;       // 0..DIN-1
    float acc = bt[col];
    const float* s = slots + row * DSLOT_;
    #pragma unroll 4
    for (int i = 0; i < DSLOT_; ++i)
        acc = fmaf(s[i], Wt[i * DIN_ + col], acc);
    sp[row * DIN_ + col] = acc;
}

// ---------------------------------------------------------------------------
// Top-4 of 16 mask values per (b,p). idxs layout (B, TOPK, P).
// Strict '>' keeps lowest index on ties (matches jax.lax.top_k).
// ---------------------------------------------------------------------------
__global__ void topk_kernel(const float* __restrict__ masks,
                            int* __restrict__ idxs) {
    int i = blockIdx.x * blockDim.x + threadIdx.x;
    if (i >= B_ * P_) return;
    int b = i >> 10;
    int p = i & (P_ - 1);
    float v[K_];
    #pragma unroll
    for (int k = 0; k < K_; ++k) v[k] = masks[(b * K_ + k) * P_ + p];
    unsigned used = 0;
    #pragma unroll
    for (int t = 0; t < TOPK_; ++t) {
        float best = -3.4e38f; int bi = 0;
        #pragma unroll
        for (int k = 0; k < K_; ++k)
            if (!((used >> k) & 1u) && v[k] > best) { best = v[k]; bi = k; }
        used |= 1u << bi;
        idxs[(b * TOPK_ + t) * P_ + p] = bi;
    }
}

__global__ void zero_kernel(float* __restrict__ p, int n) {
    int i = blockIdx.x * blockDim.x + threadIdx.x;
    if (i < n) p[i] = 0.0f;
}

// ---------------------------------------------------------------------------
// Fragment helpers
// ---------------------------------------------------------------------------
// A fragment (16x32 bf16) slice at pointer p (already offset to row/ko):
// two contiguous 16B halves at p and p+16 elements.
__device__ __forceinline__ v16bf frag_from(const bf16_t* p) {
    v8bf lo = *(const v8bf*)(p);
    v8bf hi = *(const v8bf*)(p + 16);
    return __builtin_shufflevector(lo, hi,
        0, 1, 2, 3, 4, 5, 6, 7, 8, 9, 10, 11, 12, 13, 14, 15);
}

// GEMM over an adjacent pair of N-tiles (nt0, nt0+1): per K-step,
// 4 A frags + 2 B frags -> 8 WMMAs. Pointers hoisted out of the K-loop so
// the body is loads with immediate offsets + 2 pointer increments.
__device__ __forceinline__ void gemm_pair(const bf16_t* a, int ld,
                                          const bf16_t* __restrict__ Wp,
                                          int KC, int NT, int nt0, int lane,
                                          v8f& c0, v8f& c1, v8f& c2, v8f& c3,
                                          v8f& c4, v8f& c5, v8f& c6, v8f& c7) {
    const int r  = lane & 15;
    const int ko = (lane >> 4) << 3;
    const bf16_t* ap = a + r * ld + ko;                        // M-tile 0 base
    const bf16_t* bp = Wp + (size_t)nt0 * 512 + lane * 16;     // nt0 fragment
    const int bstride = NT * 512;                              // per-kc stride
    #pragma unroll 2
    for (int kc = 0; kc < KC; ++kc) {
        v16bf b0 = *(const v16bf*)(bp);
        v16bf b1 = *(const v16bf*)(bp + 512);                  // nt0+1: +1KB
        v16bf A0 = frag_from(ap);
        v16bf A1 = frag_from(ap + 16 * ld);
        v16bf A2 = frag_from(ap + 32 * ld);
        v16bf A3 = frag_from(ap + 48 * ld);
        c0 = WMMA_BF16(A0, b0, c0);
        c1 = WMMA_BF16(A1, b0, c1);
        c2 = WMMA_BF16(A2, b0, c2);
        c3 = WMMA_BF16(A3, b0, c3);
        c4 = WMMA_BF16(A0, b1, c4);
        c5 = WMMA_BF16(A1, b1, c5);
        c6 = WMMA_BF16(A2, b1, c6);
        c7 = WMMA_BF16(A3, b1, c7);
        ap += 32;
        bp += bstride;
    }
}

// Single N-tile variant (layer-3 odd tail).
__device__ __forceinline__ void gemm_one(const bf16_t* a, int ld,
                                         const bf16_t* __restrict__ Wp,
                                         int KC, int NT, int nt, int lane,
                                         v8f& c0, v8f& c1, v8f& c2, v8f& c3) {
    const int r  = lane & 15;
    const int ko = (lane >> 4) << 3;
    const bf16_t* ap = a + r * ld + ko;
    const bf16_t* bp = Wp + (size_t)nt * 512 + lane * 16;
    const int bstride = NT * 512;
    #pragma unroll 2
    for (int kc = 0; kc < KC; ++kc) {
        v16bf bb = *(const v16bf*)(bp);
        v16bf A0 = frag_from(ap);
        v16bf A1 = frag_from(ap + 16 * ld);
        v16bf A2 = frag_from(ap + 32 * ld);
        v16bf A3 = frag_from(ap + 48 * ld);
        c0 = WMMA_BF16(A0, bb, c0);
        c1 = WMMA_BF16(A1, bb, c1);
        c2 = WMMA_BF16(A2, bb, c2);
        c3 = WMMA_BF16(A3, bb, c3);
        ap += 32;
        bp += bstride;
    }
}

__device__ __forceinline__ void store_relu_4(bf16_t* h, int ld, int col,
                                             int rbase, float bias,
                                             v8f c0, v8f c1, v8f c2, v8f c3) {
    v8f cc[4] = {c0, c1, c2, c3};
    #pragma unroll
    for (int mt = 0; mt < 4; ++mt) {
        #pragma unroll
        for (int r = 0; r < 8; ++r) {
            float v = cc[mt][r] + bias;
            v = v > 0.0f ? v : 0.0f;
            h[(mt * 16 + rbase + r) * ld + col] = (bf16_t)v;
        }
    }
}

__device__ __forceinline__ void store_out_4(bf16_t* outs, float* alphas,
                                            int col, int rbase, float bias,
                                            v8f c0, v8f c1, v8f c2, v8f c3) {
    v8f cc[4] = {c0, c1, c2, c3};
    #pragma unroll
    for (int mt = 0; mt < 4; ++mt) {
        #pragma unroll
        for (int r = 0; r < 8; ++r) {
            float v = cc[mt][r] + bias;
            int row = mt * 16 + rbase + r;
            if (col < M_)       outs[row * OUT_LD + col] = (bf16_t)v;
            else if (col == M_) alphas[row] = v;     // f32 alpha for softmax
        }
    }
}

// ---------------------------------------------------------------------------
// Fused gather + 3-layer MLP + softmax/reduce/scatter.
// One workgroup = 64 rows ordered (b,p,t): 16 positions x TOPK, fixed b.
// 16 wave32 split N-tiles in adjacent pairs; each wave keeps 8 M/N-tile
// accumulators so each K-step's 4 LDS A-fragments feed 8 WMMAs and every B
// fragment fetched from L2 feeds 4 WMMAs. One pair live at a time (unroll 1)
// to keep VGPR pressure below the spill threshold.
// ---------------------------------------------------------------------------
__global__ void __launch_bounds__(THREADS_)
fused_mlp_kernel(const float* __restrict__ sp,
                 const int*   __restrict__ idxs,
                 const float* __restrict__ pos_embed,
                 const bf16_t* __restrict__ W1p, const float* __restrict__ b1,
                 const bf16_t* __restrict__ W2p, const float* __restrict__ b2,
                 const bf16_t* __restrict__ W3p, const float* __restrict__ b3,
                 float* __restrict__ recon, float* __restrict__ masks_out) {
    __shared__ __align__(128) unsigned char smem[BUFA_BYTES + BUFB_BYTES + 768];
    bf16_t* xs   = (bf16_t*)smem;                       // 64 x XS_LD  (ph 0-1)
    bf16_t* h2   = (bf16_t*)smem;                       // 64 x H_LD   (ph 2-3)
    bf16_t* h1   = (bf16_t*)(smem + BUFA_BYTES);        // 64 x H_LD   (ph 1-2)
    bf16_t* outs = (bf16_t*)(smem + BUFA_BYTES);        // 64 x OUT_LD (ph 3-4)
    int*    rowSlot = (int*)  (smem + BUFA_BYTES + BUFB_BYTES);
    float*  alphas  = (float*)(smem + BUFA_BYTES + BUFB_BYTES + 256);
    float*  wgt     = (float*)(smem + BUFA_BYTES + BUFB_BYTES + 512);

    const int tid  = threadIdx.x;
    const int lane = tid & 31;
    const int wv   = tid >> 5;            // wave index 0..15 (wave32)
    const int T    = blockIdx.x;
    const int b    = T >> 6;              // 64 tiles per batch element
    const int p0   = (T & 63) << 4;       // first of 16 positions

    // ---- Phase 0: gather x = slot_proj[b, topk_idx] + pos_embed[p] -> bf16
    {
        int r   = tid >> 3;               // local row 0..63 ; r = 4*(p-p0)+t
        int seg = tid & 7;                // 32 elements per thread
        int t   = r & 3;
        int p   = p0 + (r >> 2);
        int slot = idxs[(b * TOPK_ + t) * P_ + p];
        if (seg == 0) rowSlot[r] = slot;
        const float* srow = sp + (size_t)(b * K_ + slot) * DIN_;
        const float* prow = pos_embed + (size_t)p * DIN_;
        int d0 = seg * 32;
        #pragma unroll
        for (int j = 0; j < 16; ++j) {
            float f0 = srow[d0 + 2 * j]     + prow[d0 + 2 * j];
            float f1 = srow[d0 + 2 * j + 1] + prow[d0 + 2 * j + 1];
            v2bf pv; pv[0] = (bf16_t)f0; pv[1] = (bf16_t)f1;
            *(v2bf*)&xs[r * XS_LD + d0 + 2 * j] = pv;
        }
    }
    __syncthreads();

    const int rbase = (lane >> 4) << 3;
    const int cl    = lane & 15;

    // ---- Phase 1: h1 = relu(xs @ W1 + b1)      (64x256 @ 256x1024)
    #pragma unroll 1
    for (int half = 0; half < 2; ++half) {
        int nt0 = half * 32 + 2 * wv;
        int nt1 = nt0 + 1;
        v8f c0 = {}, c1 = {}, c2 = {}, c3 = {}, c4 = {}, c5 = {}, c6 = {}, c7 = {};
        gemm_pair(xs, XS_LD, W1p, KC1, NT1, nt0, lane,
                  c0, c1, c2, c3, c4, c5, c6, c7);
        store_relu_4(h1, H_LD, nt0 * 16 + cl, rbase, b1[nt0 * 16 + cl], c0, c1, c2, c3);
        store_relu_4(h1, H_LD, nt1 * 16 + cl, rbase, b1[nt1 * 16 + cl], c4, c5, c6, c7);
    }
    __syncthreads();

    // ---- Phase 2: h2 = relu(h1 @ W2 + b2)      (64x1024 @ 1024x1024)
    #pragma unroll 1
    for (int half = 0; half < 2; ++half) {
        int nt0 = half * 32 + 2 * wv;
        int nt1 = nt0 + 1;
        v8f c0 = {}, c1 = {}, c2 = {}, c3 = {}, c4 = {}, c5 = {}, c6 = {}, c7 = {};
        gemm_pair(h1, H_LD, W2p, KC2, NT2, nt0, lane,
                  c0, c1, c2, c3, c4, c5, c6, c7);
        store_relu_4(h2, H_LD, nt0 * 16 + cl, rbase, b2[nt0 * 16 + cl], c0, c1, c2, c3);
        store_relu_4(h2, H_LD, nt1 * 16 + cl, rbase, b2[nt1 * 16 + cl], c4, c5, c6, c7);
    }
    __syncthreads();

    // ---- Phase 3: out = h2 @ W3 + b3           (64x1024 @ 1024x769, pad 784)
    #pragma unroll 1
    for (int half = 0; half < 2; ++half) {
        int nt0 = half * 32 + 2 * wv;
        int nt1 = nt0 + 1;
        if (nt1 < NT3) {
            v8f c0 = {}, c1 = {}, c2 = {}, c3 = {}, c4 = {}, c5 = {}, c6 = {}, c7 = {};
            gemm_pair(h2, H_LD, W3p, KC3, NT3, nt0, lane,
                      c0, c1, c2, c3, c4, c5, c6, c7);
            store_out_4(outs, alphas, nt0 * 16 + cl, rbase,
                        b3[nt0 * 16 + cl], c0, c1, c2, c3);
            store_out_4(outs, alphas, nt1 * 16 + cl, rbase,
                        b3[nt1 * 16 + cl], c4, c5, c6, c7);
        } else if (nt0 < NT3) {
            v8f c0 = {}, c1 = {}, c2 = {}, c3 = {};
            gemm_one(h2, H_LD, W3p, KC3, NT3, nt0, lane, c0, c1, c2, c3);
            int col = nt0 * 16 + cl;
            float bias = (col <= M_) ? b3[col] : 0.0f;
            store_out_4(outs, alphas, col, rbase, bias, c0, c1, c2, c3);
        }
    }
    __syncthreads();

    // ---- Phase 4: softmax over TOPK per position, weighted sum, mask scatter
    if (tid < ROWS_) {
        int g = tid >> 2, t = tid & 3;
        float a0 = alphas[4 * g + 0];
        float a1 = alphas[4 * g + 1];
        float a2 = alphas[4 * g + 2];
        float a3 = alphas[4 * g + 3];
        float mx = fmaxf(fmaxf(a0, a1), fmaxf(a2, a3));
        float e0 = __expf(a0 - mx), e1 = __expf(a1 - mx);
        float e2 = __expf(a2 - mx), e3 = __expf(a3 - mx);
        float den = e0 + e1 + e2 + e3;
        float mine = (t == 0) ? e0 : (t == 1) ? e1 : (t == 2) ? e2 : e3;
        wgt[tid] = mine / den;
    }
    __syncthreads();

    for (int i = tid; i < NPOS_ * M_; i += THREADS_) {
        int g = i / M_;
        int m = i - g * M_;
        float w0 = wgt[4 * g + 0], w1 = wgt[4 * g + 1];
        float w2 = wgt[4 * g + 2], w3 = wgt[4 * g + 3];
        float acc = (float)outs[(4 * g + 0) * OUT_LD + m] * w0
                  + (float)outs[(4 * g + 1) * OUT_LD + m] * w1
                  + (float)outs[(4 * g + 2) * OUT_LD + m] * w2
                  + (float)outs[(4 * g + 3) * OUT_LD + m] * w3;
        recon[((size_t)(b * P_ + p0 + g)) * M_ + m] = acc;
    }
    if (tid < ROWS_) {
        int g = tid >> 2, t = tid & 3;
        int slot = rowSlot[4 * g + t];
        masks_out[(size_t)(b * K_ + slot) * P_ + (p0 + g)] = wgt[tid];
    }
}

// ---------------------------------------------------------------------------
// Launch
// ---------------------------------------------------------------------------
extern "C" void kernel_launch(void* const* d_in, const int* in_sizes, int n_in,
                              void* d_out, int out_size, void* d_ws, size_t ws_size,
                              hipStream_t stream) {
    (void)in_sizes; (void)n_in; (void)out_size; (void)ws_size;

    const float* slots = (const float*)d_in[0];
    const float* masks = (const float*)d_in[1];
    const float* Wt    = (const float*)d_in[2];
    const float* bt    = (const float*)d_in[3];
    const float* pos   = (const float*)d_in[4];
    const float* W1    = (const float*)d_in[5];
    const float* b1    = (const float*)d_in[6];
    const float* W2    = (const float*)d_in[7];
    const float* b2    = (const float*)d_in[8];
    const float* W3    = (const float*)d_in[9];
    const float* b3    = (const float*)d_in[10];

    // Workspace layout
    char* ws = (char*)d_ws;
    float*  sp   = (float*)(ws + 0);                 // 512*256 f32   = 512 KB
    int*    idxs = (int*)  (ws + (512u * 1024u));    // 32*4*1024 i32 = 512 KB
    bf16_t* W1p  = (bf16_t*)(ws + (1024u * 1024u));  // 512 KB
    bf16_t* W2p  = (bf16_t*)(ws + (1536u * 1024u));  // 2 MB
    bf16_t* W3p  = (bf16_t*)(ws + (3584u * 1024u));  // ~1.57 MB

    float* recon     = (float*)d_out;
    float* masks_out = recon + (size_t)B_ * P_ * M_;

    // Pre-passes (all independent, serialized on stream)
    {
        int tot1 = KC1 * NT1 * 512;
        pack_weights_kernel<<<(tot1 + 255) / 256, 256, 0, stream>>>(
            W1, W1p, DIN_, H_, NT1, tot1);
        int tot2 = KC2 * NT2 * 512;
        pack_weights_kernel<<<(tot2 + 255) / 256, 256, 0, stream>>>(
            W2, W2p, H_, H_, NT2, tot2);
        int tot3 = KC3 * NT3 * 512;
        pack_weights_kernel<<<(tot3 + 255) / 256, 256, 0, stream>>>(
            W3, W3p, H_, M_ + 1, NT3, tot3);
    }
    slot_proj_kernel<<<B_ * K_, DIN_, 0, stream>>>(slots, Wt, bt, sp);
    topk_kernel<<<(B_ * P_ + 255) / 256, 256, 0, stream>>>(masks, idxs);
    zero_kernel<<<(B_ * K_ * P_ + 255) / 256, 256, 0, stream>>>(
        masks_out, B_ * K_ * P_);

    // Fused compute: 2048 tiles of 64 rows, 512 threads (16 wave32) each
    fused_mlp_kernel<<<B_ * (P_ / NPOS_), THREADS_, 0, stream>>>(
        sp, idxs, pos, W1p, b1, W2p, b2, W3p, b3, recon, masks_out);
}
